// mDACF_89833535963795
// MI455X (gfx1250) — matrix operand order repaired and checked by reading, not compiled, not run on hardware
//
#include <hip/hip_runtime.h>

// CDNA5 (gfx1250) wave32 WMMA f16 path, double-buffered LDS pipeline.
typedef __attribute__((ext_vector_type(16))) _Float16 v16h;
typedef __attribute__((ext_vector_type(8)))  _Float16 v8h;
typedef __attribute__((ext_vector_type(4)))  _Float16 v4h;
typedef __attribute__((ext_vector_type(8)))  float    v8f;

#define LATENT      128
#define DFEAT       1000
#define KPAD        1024     // zero-padded K
#define KC          32       // K-chunk = one 16x16x32 WMMA step
#define NCHUNK      (KPAD / KC)
#define LDS_STRIDE  40       // halfs per row (KC + 8 pad -> bank-conflict break)
#define ROWS_PER_WG 64       // 4 waves x 16-row M tiles -> 256 workgroups
#define NWAVES      4
#define TILE_M      16
#define BATCH       16384

__launch_bounds__(128)
__global__ void twotower_diag_kernel(const int* __restrict__ x,
                                     const float* __restrict__ user_lookup,
                                     const float* __restrict__ item_lookup,
                                     const float* __restrict__ uWenc,
                                     const float* __restrict__ ubenc,
                                     const float* __restrict__ iWenc,
                                     const float* __restrict__ ibenc,
                                     float* __restrict__ out)
{
    // double-buffered staging tiles (f16), ~60 KB total
    __shared__ __align__(16) _Float16 Au[2][ROWS_PER_WG * LDS_STRIDE];
    __shared__ __align__(16) _Float16 Ai[2][ROWS_PER_WG * LDS_STRIDE];
    __shared__ __align__(16) _Float16 Bu[2][LATENT * LDS_STRIDE];   // W chunk transposed [n][k]
    __shared__ __align__(16) _Float16 Bi[2][LATENT * LDS_STRIDE];

    const int tid     = threadIdx.x;
    const int lane    = tid & 31;
    const int wave    = tid >> 5;                 // 0..3
    const int rowBase = blockIdx.x * ROWS_PER_WG;

    // ---- staging roles (128 threads) ----
    const int sRow  = tid >> 1;                   // A staging: row 0..63
    const int sHalf = tid & 1;                    // A staging: 16-f32 half of 32-K chunk
    const int wCol  = tid;                        // B staging: latent column n (0..127)

    const int uIdx = x[(rowBase + sRow) * 2 + 0];
    const int iIdx = x[(rowBase + sRow) * 2 + 1];
    const float* __restrict__ uRow = user_lookup + (size_t)uIdx * DFEAT;
    const float* __restrict__ iRow = item_lookup + (size_t)iIdx * DFEAT;

    // ---- WMMA fragment lane mapping (wave32; A and B both use contiguous k = sel*16 + e) ----
    const int mLane = lane & 15;                  // M (A) / N (B,C,D) within tile
    const int sel   = lane >> 4;                  // half-wave selector
    const int aOff  = (wave * TILE_M + mLane) * LDS_STRIDE + sel * 16;

    // staging prefetch registers
    float4 pfU[4], pfI[4];
    float  pwU[KC], pwI[KC];

    auto stageLoad = [&](int k0) {
#pragma unroll
        for (int j = 0; j < 4; ++j) {             // A rows: guarded float4 gathers
            const int koff = sHalf * 16 + j * 4;
            const int gk   = k0 + koff;           // DFEAT % 4 == 0
            pfU[j] = make_float4(0.f, 0.f, 0.f, 0.f);
            pfI[j] = pfU[j];
            if (gk < DFEAT) {
                pfU[j] = *(const float4*)(uRow + gk);
                pfI[j] = *(const float4*)(iRow + gk);
            }
        }
#pragma unroll
        for (int k = 0; k < KC; ++k) {            // W columns: coalesced across lanes
            const int gk = k0 + k;
            pwU[k] = 0.f; pwI[k] = 0.f;
            if (gk < DFEAT) {
                pwU[k] = uWenc[(size_t)gk * LATENT + wCol];
                pwI[k] = iWenc[(size_t)gk * LATENT + wCol];
            }
        }
    };
    auto stageStore = [&](int buf) {
#pragma unroll
        for (int j = 0; j < 4; ++j) {             // f32 -> f16 convert + 8B LDS stores
            const int koff = sHalf * 16 + j * 4;
            v4h hu = { (_Float16)pfU[j].x, (_Float16)pfU[j].y, (_Float16)pfU[j].z, (_Float16)pfU[j].w };
            v4h hi = { (_Float16)pfI[j].x, (_Float16)pfI[j].y, (_Float16)pfI[j].z, (_Float16)pfI[j].w };
            *(v4h*)(&Au[buf][sRow * LDS_STRIDE + koff]) = hu;
            *(v4h*)(&Ai[buf][sRow * LDS_STRIDE + koff]) = hi;
        }
#pragma unroll
        for (int k = 0; k < KC; k += 4) {
            v4h hu = { (_Float16)pwU[k], (_Float16)pwU[k+1], (_Float16)pwU[k+2], (_Float16)pwU[k+3] };
            v4h hi = { (_Float16)pwI[k], (_Float16)pwI[k+1], (_Float16)pwI[k+2], (_Float16)pwI[k+3] };
            *(v4h*)(&Bu[buf][wCol * LDS_STRIDE + k]) = hu;
            *(v4h*)(&Bi[buf][wCol * LDS_STRIDE + k]) = hi;
        }
    };

    v8f accU[8], accI[8];
#pragma unroll
    for (int t = 0; t < 8; ++t) { accU[t] = (v8f)(0.0f); accI[t] = (v8f)(0.0f); }

    // prologue: fill buffer 0
    stageLoad(0);
    stageStore(0);
    __syncthreads();

#pragma unroll 1
    for (int c = 0; c < NCHUNK; ++c) {
        const int  cur  = c & 1;
        const bool more = (c + 1) < NCHUNK;

        // 1) issue next chunk's global loads early (latency hides behind WMMAs)
        if (more) stageLoad((c + 1) * KC);

        // 2) one 16x16x32 K-step for both towers from current buffer
        v16h aU, aI;
        {
            const _Float16* au = &Au[cur][aOff];
            const _Float16* ai = &Ai[cur][aOff];
            v8h q0 = *(const v8h*)(au);
            v8h q1 = *(const v8h*)(au + 8);
            v8h p0 = *(const v8h*)(ai);
            v8h p1 = *(const v8h*)(ai + 8);
#pragma unroll
            for (int e = 0; e < 8; ++e) {
                aU[e] = q0[e]; aU[e + 8] = q1[e];
                aI[e] = p0[e]; aI[e + 8] = p1[e];
            }
        }
#pragma unroll
        for (int t = 0; t < 8; ++t) {
            const _Float16* bup = &Bu[cur][(t * 16 + mLane) * LDS_STRIDE + sel * 16];
            const _Float16* bip = &Bi[cur][(t * 16 + mLane) * LDS_STRIDE + sel * 16];
            v8h b0 = *(const v8h*)(bup);
            v8h b1 = *(const v8h*)(bup + 8);
            v8h c0 = *(const v8h*)(bip);
            v8h c1 = *(const v8h*)(bip + 8);
            v16h bU, bI;
#pragma unroll
            for (int e = 0; e < 8; ++e) {
                bU[e] = b0[e]; bU[e + 8] = b1[e];
                bI[e] = c0[e]; bI[e + 8] = c1[e];
            }
            accU[t] = __builtin_amdgcn_wmma_f32_16x16x32_f16(
                false, aU, false, bU, (short)0, accU[t], false, false);
            accI[t] = __builtin_amdgcn_wmma_f32_16x16x32_f16(
                false, aI, false, bI, (short)0, accI[t], false, false);
        }

        // 3) commit next chunk into the alternate buffer (last read in the previous iter, before its barrier -> safe)
        if (more) stageStore(cur ^ 1);

        // 4) single barrier per chunk
        __syncthreads();
    }

    // ---- epilogue: bias + tanh + diagonal dot ----
    // C/D layout: lane n (=mLane) VGPR j = Z[M = sel*8 + j][N = t*16 + n]
    float s[8];
#pragma unroll
    for (int j = 0; j < 8; ++j) s[j] = 0.f;
#pragma unroll
    for (int t = 0; t < 8; ++t) {
        const float bu = ubenc[t * 16 + mLane];
        const float bi = ibenc[t * 16 + mLane];
#pragma unroll
        for (int j = 0; j < 8; ++j) {
            const float zu = tanhf(accU[t][j] + bu);
            const float zi = tanhf(accI[t][j] + bi);
            s[j] += zu * zi;
        }
    }
    // reduce over N across the 16 lanes of each half-wave (masks < 16 stay in-half)
#pragma unroll
    for (int m = 1; m < 16; m <<= 1) {
#pragma unroll
        for (int j = 0; j < 8; ++j) s[j] += __shfl_xor(s[j], m, 32);
    }
    if (mLane == 0) {
        const int base = rowBase + wave * TILE_M + sel * 8;
#pragma unroll
        for (int j = 0; j < 8; ++j) out[base + j] = s[j];
    }
}

extern "C" void kernel_launch(void* const* d_in, const int* in_sizes, int n_in,
                              void* d_out, int out_size, void* d_ws, size_t ws_size,
                              hipStream_t stream) {
    (void)in_sizes; (void)n_in; (void)d_ws; (void)ws_size; (void)out_size;
    // setup_inputs order:
    // 0:x 1:user_lookup 2:item_lookup 3:user_W_enc 4:user_b_enc 5:user_W_dec
    // 6:user_b_dec 7:item_W_enc 8:item_b_enc 9:item_W_dec 10:item_b_dec
    const int*   x  = (const int*)  d_in[0];
    const float* ul = (const float*)d_in[1];
    const float* il = (const float*)d_in[2];
    const float* uW = (const float*)d_in[3];
    const float* ub = (const float*)d_in[4];
    const float* iW = (const float*)d_in[7];
    const float* ib = (const float*)d_in[8];
    float* out = (float*)d_out;

    dim3 grid(BATCH / ROWS_PER_WG);   // 256 workgroups -> covers a big-WGP-count chip
    dim3 block(128);                  // 4 wave32 waves
    twotower_diag_kernel<<<grid, block, 0, stream>>>(x, ul, il, uW, ub, iW, ib, out);
}
